// AnnCloseModel_26860725469415
// MI455X (gfx1250) — matrix operand order repaired and checked by reading, not compiled, not run on hardware
//
#include <hip/hip_runtime.h>

typedef __attribute__((ext_vector_type(16))) _Float16 v16h;
typedef __attribute__((ext_vector_type(8)))  _Float16 v8h;
typedef __attribute__((ext_vector_type(8)))  float    v8f;
typedef __attribute__((ext_vector_type(4)))  float    v4f;

#define NT_    365
#define NGRID_ 30000
#define NX_    16
#define HID_   256
#define TILE_M 16
#define NWAVES 16          // 512 threads; one 16-wide N-tile per wave

__device__ __forceinline__ v8f wmma_f16(v16h a, v16h b, v8f c) {
  // (neg_a, A, neg_b, B, c_mod, C, reuse_a, reuse_b)
  return __builtin_amdgcn_wmma_f32_16x16x32_f16(false, a, false, b, (short)0, c, false, false);
}

// A-fragment (16x32 f16, M x K) in this code's contiguous k order:
// lane: M = lane%16; K = half*16 + [0..15]
__device__ __forceinline__ v16h load_a_frag(const _Float16* row, int kbase, int half) {
  const v8h* p0 = (const v8h*)(row + kbase + half * 16);
  const v8h* p1 = (const v8h*)(row + kbase + half * 16 + 8);
  v8h a0 = *p0;
  v8h a1 = *p1;
  v16h a;
#pragma unroll
  for (int i = 0; i < 8; ++i) { a[i] = a0[i]; a[8 + i] = a1[i]; }
  return a;
}

__global__ __launch_bounds__(32 * NWAVES)
void rnn_closedloop_wmma(const float* __restrict__ x,  const float* __restrict__ y,
                         const float* __restrict__ Wi, const float* __restrict__ bi,
                         const float* __restrict__ Wh, const float* __restrict__ bh,
                         const float* __restrict__ Wo, const float* __restrict__ bo,
                         float* __restrict__ out)
{
  __shared__ _Float16 xt_lds[TILE_M * 32];       // 16 rows x K=32 (17 real + pad)
  __shared__ _Float16 ht_lds[TILE_M * HID_];     // 16 x 256 f16 = 8KB
  __shared__ float    wave_part[NWAVES][TILE_M];
  __shared__ float    yt_state[TILE_M];

  const int tid  = threadIdx.x;
  const int wave = tid >> 5;          // 0..15 == N-tile index
  const int lane = tid & 31;
  const int half = lane >> 4;
  const int l16  = lane & 15;
  const int row0 = blockIdx.x * TILE_M;

  // ---- Load this wave's weight B-fragments into registers once (365-step reuse) ----
  // B layout (K x 16, f16): lane N = lane%16; K = (lane/16)*16 + [0..15] over 8 VGPR pairs
  v16h bwi;
  v16h bwh[8];
  const int col = wave * 16 + l16;    // output hidden unit = row of Wi/Wh, col of Wo
  // Wi^T fragment: K = 17 zero-padded to 32
#pragma unroll
  for (int u = 0; u < 16; ++u) {
    const int k = half * 16 + u;
    bwi[u] = (k < NX_ + 1) ? (_Float16)Wi[col * (NX_ + 1) + k] : (_Float16)0.0f;
  }
  // Wh^T fragments: 8 K-chunks of 32; per-lane 16 contiguous floats each
  {
    const float* wrow = Wh + (size_t)col * HID_;
#pragma unroll
    for (int kc = 0; kc < 8; ++kc) {
      const v4f* p = (const v4f*)(wrow + kc * 32 + half * 16);
#pragma unroll
      for (int q = 0; q < 4; ++q) {
        v4f f = p[q];
#pragma unroll
        for (int e = 0; e < 4; ++e) bwh[kc][q * 4 + e] = (_Float16)f[e];
      }
    }
  }
  const float bi_l = bi[col];
  const float bh_l = bh[col];
  const float wo_l = Wo[col];
  const float bo0  = bo[0];

  // Zero the padded K columns 16..31 once; init recurrent state yt = 0.
  if (tid < 256) {
    const int r = tid >> 4, c = tid & 15;
    xt_lds[r * 32 + 16 + c] = (_Float16)0.0f;
    if (tid < TILE_M) yt_state[tid] = 0.0f;
  }

  for (int t = 0; t < NT_; ++t) {
    __syncthreads();   // yt_state ready; previous readers of xt/ht done

    // ---- Build xt tile: 16x16 x-features (coalesced 1KB) + fillObs column ----
    if (tid < 256) {
      const int r = tid >> 4, f = tid & 15;
      const float xv = x[((size_t)t * NGRID_ + row0 + r) * NX_ + f];
      xt_lds[r * 32 + f] = (_Float16)xv;
      if (tid < TILE_M) {
        const float yv   = y[(size_t)t * NGRID_ + row0 + tid];
        const float fill = (yv != yv) ? yt_state[tid] : yv;   // NaN -> closed loop
        xt_lds[tid * 32 + 16] = (_Float16)fill;
      }
      if (t + 1 < NT_ && tid < 9) {   // prefetch next step's x tile (8 x 128B) + y line
        if (tid < 8) {
          const float* nx = x + ((size_t)(t + 1) * NGRID_ + row0) * NX_;
          __builtin_prefetch(nx + tid * 32, 0, 3);
        } else {
          const float* ny = y + (size_t)(t + 1) * NGRID_ + row0;
          __builtin_prefetch(ny, 0, 3);
        }
      }
    }
    __syncthreads();

    // ---- Layer 1: ht = relu(xt @ Wi^T + bi); one WMMA per wave ----
    {
      const v16h a = load_a_frag(xt_lds + l16 * 32, 0, half);
      v8f acc = {};
      acc = wmma_f16(a, bwi, acc);
#pragma unroll
      for (int r = 0; r < 8; ++r) {            // C layout: M = r + 8*half, N = wave*16+l16
        float v = acc[r] + bi_l;
        v = v > 0.0f ? v : 0.0f;
        ht_lds[(r + 8 * half) * HID_ + wave * 16 + l16] = (_Float16)v;
      }
    }
    __syncthreads();

    // ---- Layer 2: ht2 = ht @ Wh^T + bh (8 chained WMMAs per wave), then Wo dot ----
    {
      // Hoist all A-fragment LDS loads ahead of the WMMA chain so the in-order
      // DS returns overlap the serial accumulate instead of stalling per chunk.
      v16h afrag[8];
#pragma unroll
      for (int kc = 0; kc < 8; ++kc)
        afrag[kc] = load_a_frag(ht_lds + l16 * HID_, kc * 32, half);

      v8f acc;
#pragma unroll
      for (int r = 0; r < 8; ++r) acc[r] = bh_l;
#pragma unroll
      for (int kc = 0; kc < 8; ++kc)
        acc = wmma_f16(afrag[kc], bwh[kc], acc);

      // Output head: per-lane partial over this lane's N, reduce across 16-lane half
      float val[8];
#pragma unroll
      for (int r = 0; r < 8; ++r) val[r] = acc[r] * wo_l;
#pragma unroll
      for (int off = 8; off >= 1; off >>= 1) {
#pragma unroll
        for (int r = 0; r < 8; ++r) val[r] += __shfl_xor(val[r], off, 32);
      }
      if (l16 == 0) {
#pragma unroll
        for (int r = 0; r < 8; ++r) wave_part[wave][8 * half + r] = val[r];
      }
    }
    __syncthreads();

    // ---- Final cross-wave reduce, emit yt, update recurrent state ----
    if (tid < TILE_M) {
      float s = bo0;
#pragma unroll
      for (int w = 0; w < NWAVES; ++w) s += wave_part[w][tid];
      out[(size_t)t * NGRID_ + row0 + tid] = s;
      yt_state[tid] = s;
    }
  }
}

extern "C" void kernel_launch(void* const* d_in, const int* in_sizes, int n_in,
                              void* d_out, int out_size, void* d_ws, size_t ws_size,
                              hipStream_t stream) {
  const float* x  = (const float*)d_in[0];
  const float* y  = (const float*)d_in[1];
  const float* Wi = (const float*)d_in[2];
  const float* bi = (const float*)d_in[3];
  const float* Wh = (const float*)d_in[4];
  const float* bh = (const float*)d_in[5];
  const float* Wo = (const float*)d_in[6];
  const float* bo = (const float*)d_in[7];
  float* out = (float*)d_out;

  rnn_closedloop_wmma<<<NGRID_ / TILE_M, 32 * NWAVES, 0, stream>>>(x, y, Wi, bi, Wh, bh, Wo, bo, out);
}
